// GQADecodeKernel_63702954934391
// MI455X (gfx1250) — matrix operand order repaired and checked by reading, not compiled, not run on hardware
//
#include <hip/hip_runtime.h>

typedef __attribute__((ext_vector_type(16))) __bf16 v16bf;
typedef __attribute__((ext_vector_type(2)))  __bf16 v2bf;
typedef __attribute__((ext_vector_type(8)))  float  v8f;
typedef __attribute__((ext_vector_type(4)))  float  v4f;

namespace {
constexpr int kB = 8;
constexpr int kH = 32;
constexpr int kS = 8192;
constexpr int kG = 8;
constexpr int kD = 128;
constexpr int kGQ = kH / kG;               // 4 query heads per kv group
constexpr int kPartStride = 8 + kGQ * kD;  // floats per partial record (520)
constexpr float kNegInf = -1e30f;
constexpr float kScale = 0.08838834764831845f;  // 1/sqrt(128)
}

// Broadcast a float from a compile-time-constant lane via v_readlane_b32 (SGPR
// result -> feeds v_fmac_f32 as a scalar operand, no LDS traffic).
__device__ __forceinline__ float readlane_f(float v, int srclane) {
  return __builtin_bit_cast(
      float, __builtin_amdgcn_readlane(__builtin_bit_cast(int, v), srclane));
}

// Pack two floats into one dword of bf16 pairs -> single v_cvt_pk_bf16_f32.
__device__ __forceinline__ v2bf pk2(float x, float y) {
  v2bf r;
  r[0] = (__bf16)x;
  r[1] = (__bf16)y;
  return r;
}

union V16U {
  v16bf v;
  v2bf d[8];
};

// ---------------------------------------------------------------------------
// Kernel 1: one wave32 per (b, g, partition). Streams kS/nPart keys with
// online softmax. Q.K^T via v_wmma_f32_16x16x32_bf16; attn.V via coalesced
// non-temporal b128 loads + VALU FMA with readlane-broadcast weights.
// ---------------------------------------------------------------------------
__global__ __launch_bounds__(32) void gqa_partial_kernel(
    const float* __restrict__ q, const float* __restrict__ k,
    const float* __restrict__ v, const int* __restrict__ mask,
    float* __restrict__ ws) {
  const int lane = threadIdx.x;  // 0..31 (wave32)
  const int part = blockIdx.x;
  const int g = blockIdx.y;
  const int b = blockIdx.z;
  const int npart = (int)gridDim.x;
  const int chunk = kS / npart;
  const int ntiles = chunk >> 4;

  const int col = lane & 15;   // head column (valid < kGQ) / key row in tile
  const int half = lane >> 4;  // which 16-lane half

  // ---- Build scale*Q as WMMA B-operand (dims x 16 cols), bf16 ----
  // B layout (16-bit, 32x16): lane half h, elements 0..7 = K 8h..8h+7,
  // elements 8..15 = K 16+8h..; column = lane&15 = query head.
  // The softmax scale and the zeroing of the 12 unused columns are folded
  // into a single per-lane multiplier.
  const int qhead = (col < kGQ) ? col : 0;
  const float qmul = (col < kGQ) ? kScale : 0.0f;
  const float* qrow = q + ((size_t)b * kH + (size_t)g * kGQ + qhead) * kD;
  v16bf qb[4];
#pragma unroll
  for (int c = 0; c < 4; ++c) {
    const int dbase = 32 * c + 8 * half;
    const v4f q0 = *(const v4f*)(qrow + dbase);
    const v4f q1 = *(const v4f*)(qrow + dbase + 4);
    const v4f q2 = *(const v4f*)(qrow + dbase + 16);
    const v4f q3 = *(const v4f*)(qrow + dbase + 20);
    V16U u;
    u.d[0] = pk2(q0[0] * qmul, q0[1] * qmul);
    u.d[1] = pk2(q0[2] * qmul, q0[3] * qmul);
    u.d[2] = pk2(q1[0] * qmul, q1[1] * qmul);
    u.d[3] = pk2(q1[2] * qmul, q1[3] * qmul);
    u.d[4] = pk2(q2[0] * qmul, q2[1] * qmul);
    u.d[5] = pk2(q2[2] * qmul, q2[3] * qmul);
    u.d[6] = pk2(q3[0] * qmul, q3[1] * qmul);
    u.d[7] = pk2(q3[2] * qmul, q3[3] * qmul);
    qb[c] = u.v;
  }

  // Online-softmax state. m/l tracked per lane for head `col` (lanes n and
  // n+16 hold identical copies). acc: lane holds dims 4*lane..4*lane+3 for
  // each of the 4 heads.
  float m_mine = kNegInf;
  float l_mine = 0.0f;
  v4f acc0 = {0.f, 0.f, 0.f, 0.f};
  v4f acc1 = {0.f, 0.f, 0.f, 0.f};
  v4f acc2 = {0.f, 0.f, 0.f, 0.f};
  v4f acc3 = {0.f, 0.f, 0.f, 0.f};

  const size_t rowstride = (size_t)kG * kD;  // 1024 floats between keys
  const float* kbase = k + (size_t)b * kS * rowstride + (size_t)g * kD;
  const float* vbase = v + (size_t)b * kS * rowstride + (size_t)g * kD;
  const int* mbase = mask + (size_t)b * kS * kG + g;

  const int sstart = part * chunk;

  for (int t = 0; t < ntiles; ++t) {
    const int s0 = sstart + t * 16;

    // Ballot the tile's mask: bit i = mask of key i (lanes 16-31 duplicate
    // keys 0-15, so bits 16-31 mirror bits 0-15).
    const int mymask = mbase[(size_t)(s0 + col) * kG];
    const unsigned bal = __builtin_amdgcn_ballot_w32(mymask != 0);
    const unsigned balh = bal >> (8 * half);  // align this half's keys to bit 0

    // ---- Load K tile as WMMA A-operand (16 keys x 32 dims per chunk) ----
    // A layout (16-bit, 16x32): lane m(+16h) row=key m, elements 0..7 = dims
    // 32c+8h.., elements 8..15 = dims 32c+16+8h..
    const float* krow = kbase + (size_t)(s0 + col) * rowstride;
    __builtin_prefetch(krow + 16 * rowstride, 0, 1);  // next tile into L2
    v16bf a[4];
#pragma unroll
    for (int c = 0; c < 4; ++c) {
      const int b0 = 32 * c + 8 * half;
      const int b1 = b0 + 16;
      const v4f f0 = __builtin_nontemporal_load((const v4f*)(krow + b0));
      const v4f f1 = __builtin_nontemporal_load((const v4f*)(krow + b0 + 4));
      const v4f f2 = __builtin_nontemporal_load((const v4f*)(krow + b1));
      const v4f f3 = __builtin_nontemporal_load((const v4f*)(krow + b1 + 4));
      V16U u;
      u.d[0] = pk2(f0[0], f0[1]);
      u.d[1] = pk2(f0[2], f0[3]);
      u.d[2] = pk2(f1[0], f1[1]);
      u.d[3] = pk2(f1[2], f1[3]);
      u.d[4] = pk2(f2[0], f2[1]);
      u.d[5] = pk2(f2[2], f2[3]);
      u.d[6] = pk2(f3[0], f3[1]);
      u.d[7] = pk2(f3[2], f3[3]);
      a[c] = u.v;
    }

    // ---- Scores: D += K_tile * (scale*Q)^T over the 128-dim inner product ----
    v8f sc = {0.f, 0.f, 0.f, 0.f, 0.f, 0.f, 0.f, 0.f};
    sc = __builtin_amdgcn_wmma_f32_16x16x32_bf16(false, a[0], false, qb[0],
                                                 (short)0, sc, false, false);
    sc = __builtin_amdgcn_wmma_f32_16x16x32_bf16(false, a[1], false, qb[1],
                                                 (short)0, sc, false, false);
    sc = __builtin_amdgcn_wmma_f32_16x16x32_bf16(false, a[2], false, qb[2],
                                                 (short)0, sc, false, false);
    sc = __builtin_amdgcn_wmma_f32_16x16x32_bf16(false, a[3], false, qb[3],
                                                 (short)0, sc, false, false);

    // ---- Mask bias + online softmax in C-layout ----
    // C layout: VGPR r, lane n<16 -> (key r, head n); lane n+16 -> (key r+8).
    float val[8];
    float tmax = kNegInf;
#pragma unroll
    for (int r = 0; r < 8; ++r) {
      const float bias = (balh & (1u << r)) ? 0.0f : kNegInf;
      const float x = sc[r] + bias;
      val[r] = x;
      tmax = fmaxf(tmax, x);
    }
    tmax = fmaxf(tmax, __shfl_xor(tmax, 16, 32));  // combine key halves

    const float m_new = fmaxf(m_mine, tmax);
    const float alpha = __expf(m_mine - m_new);
    m_mine = m_new;

    float p[8];
    float psum = 0.0f;
#pragma unroll
    for (int r = 0; r < 8; ++r) {
      p[r] = __expf(val[r] - m_new);
      psum += p[r];
    }
    psum += __shfl_xor(psum, 16, 32);
    l_mine = l_mine * alpha + psum;

    // Broadcast per-head rescale factors (lanes 0..3 hold heads 0..3).
    const float a0 = readlane_f(alpha, 0);
    const float a1 = readlane_f(alpha, 1);
    const float a2 = readlane_f(alpha, 2);
    const float a3 = readlane_f(alpha, 3);
    acc0 *= a0;
    acc1 *= a1;
    acc2 *= a2;
    acc3 *= a3;

    // ---- attn . V : coalesced non-temporal b128 row loads; p broadcast via
    // v_readlane (SGPR operand into v_fmac). ----
    const float* vrow = vbase + (size_t)s0 * rowstride + 4 * lane;
    __builtin_prefetch(vrow + 16 * rowstride, 0, 1);
#pragma unroll
    for (int kk = 0; kk < 16; ++kk) {
      const v4f vv = __builtin_nontemporal_load(
          (const v4f*)(vrow + (size_t)kk * rowstride));
      const int srcl = (kk >> 3) << 4;  // which lane half holds key kk
      const float p0 = readlane_f(p[kk & 7], 0 + srcl);
      const float p1 = readlane_f(p[kk & 7], 1 + srcl);
      const float p2 = readlane_f(p[kk & 7], 2 + srcl);
      const float p3 = readlane_f(p[kk & 7], 3 + srcl);
      acc0 += p0 * vv;
      acc1 += p1 * vv;
      acc2 += p2 * vv;
      acc3 += p3 * vv;
    }
  }

  // ---- Store partial record: m[4], l[4], acc[4][128] ----
  float* wp = ws + (size_t)((b * kG + g) * npart + part) * kPartStride;
  if (lane < kGQ) {
    wp[lane] = m_mine;
    wp[4 + lane] = l_mine;
  }
  float* wacc = wp + 8;
  *(v4f*)(wacc + 0 * kD + 4 * lane) = acc0;
  *(v4f*)(wacc + 1 * kD + 4 * lane) = acc1;
  *(v4f*)(wacc + 2 * kD + 4 * lane) = acc2;
  *(v4f*)(wacc + 3 * kD + 4 * lane) = acc3;
}

// ---------------------------------------------------------------------------
// Kernel 2: combine the npart partials per (b, head) and normalize.
// ---------------------------------------------------------------------------
__global__ __launch_bounds__(kD) void gqa_reduce_kernel(
    const float* __restrict__ ws, float* __restrict__ out, int npart) {
  const int hglob = blockIdx.x;  // 0..H-1, hglob = g*kGQ + hh
  const int b = blockIdx.y;
  const int g = hglob / kGQ;
  const int hh = hglob % kGQ;
  const int d = threadIdx.x;  // 0..127

  const float* base = ws + (size_t)(b * kG + g) * npart * kPartStride;

  float M = kNegInf;
  for (int p = 0; p < npart; ++p) {
    M = fmaxf(M, base[(size_t)p * kPartStride + hh]);
  }
  float num = 0.0f;
  float den = 0.0f;
  for (int p = 0; p < npart; ++p) {
    const float* wp = base + (size_t)p * kPartStride;
    const float w = __expf(wp[hh] - M);
    den += w * wp[4 + hh];
    num += w * wp[8 + hh * kD + d];
  }
  out[((size_t)b * kH + hglob) * kD + d] = num / den;
}

extern "C" void kernel_launch(void* const* d_in, const int* in_sizes, int n_in,
                              void* d_out, int out_size, void* d_ws,
                              size_t ws_size, hipStream_t stream) {
  const float* q = (const float*)d_in[0];
  const float* k = (const float*)d_in[1];
  const float* v = (const float*)d_in[2];
  const int* mask = (const int*)d_in[3];
  float* out = (float*)d_out;
  float* ws = (float*)d_ws;

  // Pick the S-partition count: 64 (4096 waves) if the workspace allows,
  // halving until the partial records fit. Pure function of ws_size ->
  // deterministic across calls / graph replays.
  int npart = 64;
  while (npart > 4 &&
         (size_t)kB * kG * npart * kPartStride * sizeof(float) > ws_size) {
    npart >>= 1;
  }

  dim3 grid1(npart, kG, kB);  // single-wave workgroups
  gqa_partial_kernel<<<grid1, dim3(32), 0, stream>>>(q, k, v, mask, ws);

  dim3 grid2(kH, kB);  // one output row (b, head) per block
  gqa_reduce_kernel<<<grid2, dim3(kD), 0, stream>>>(ws, out, npart);
}
